// Net_89893665505528
// MI455X (gfx1250) — compile-verified, matrix operand order checked by reading, not run on hardware
//
#include <hip/hip_runtime.h>

typedef __attribute__((ext_vector_type(16))) _Float16 v16h;
typedef __attribute__((ext_vector_type(8)))  _Float16 v8h;
typedef __attribute__((ext_vector_type(8)))  float    v8f;
typedef __attribute__((ext_vector_type(4)))  float    v4f;

// Workspace layout (units: _Float16 elements)
static constexpr size_t W1_OFF = 0;         // 32*192   = 6144
static constexpr size_t W2_OFF = 6144;      // 64*288   = 18432
static constexpr size_t W3_OFF = 24576;     // 64*576   = 36864
static constexpr size_t W4_OFF = 61440;     // 512*960  = 491520
static constexpr size_t X2_OFF = 552960;    // 4096*32*165 = 21626880
static constexpr size_t X4_OFF = 22179840;  // 4096*64*35  = 9175040
static constexpr size_t X6_OFF = 31354880;  // 4096*960    = 3932160
static constexpr size_t H_OFF  = 35287040;  // 4096*512    = 2097152
// total = 37384192 f16 = ~74.8 MB

__device__ __forceinline__ v16h load_bfrag(const _Float16* p) {
    // p points at (row, k0 + kh); rows are 16B-aligned, k0 multiple of 32, kh in {0,8}
    v8h lo = *(const v8h*)p;
    v8h hi = *(const v8h*)(p + 16);
    return __builtin_shufflevector(lo, hi, 0,1,2,3,4,5,6,7,8,9,10,11,12,13,14,15);
}

__device__ __forceinline__ v8f wmma_f16(v16h a, v16h b, v8f c) {
    return __builtin_amdgcn_wmma_f32_16x16x32_f16(false, a, false, b, (short)0, c, false, false);
}

// ---------------- init: zero d_out, convert weights fp32 -> f16 ----------------
__global__ __launch_bounds__(256) void k_init(const float* __restrict__ w1,
                                              const float* __restrict__ w2,
                                              const float* __restrict__ w3,
                                              const float* __restrict__ w4,
                                              _Float16* __restrict__ wsh,
                                              float* __restrict__ out) {
    int i = blockIdx.x * 256 + threadIdx.x;
    if (i < 8407)   out[i] = 0.0f;
    if (i < 6144)   wsh[W1_OFF + i] = (_Float16)w1[i];
    if (i < 18432)  wsh[W2_OFF + i] = (_Float16)w2[i];
    if (i < 36864)  wsh[W3_OFF + i] = (_Float16)w3[i];
    if (i < 491520) wsh[W4_OFF + i] = (_Float16)w4[i];
}

// ---------------- conv1: [3,48,64] -(k8 s4)-> [32,11,15], M=165 N=32 K=192 -----
// A-fragment trick: k0+kh is a multiple of 8, so elements 0..7 (and 8..15) of a
// fragment are one full contiguous kx-row of the 8x8 window -> two 32B LDS reads.
__global__ __launch_bounds__(256) void k_conv1(const float* __restrict__ x,
                                               const float* __restrict__ b1,
                                               const _Float16* __restrict__ w1h,
                                               _Float16* __restrict__ x2,
                                               float* __restrict__ ave1) {
    __shared__ float xs[3 * 48 * 64];   // 9216 floats, 36 KB
    __shared__ float zpad[8];           // zero source for padded M rows
    __shared__ float ave_s[176];
    const int b = blockIdx.x, tid = threadIdx.x;

    {   // cooperative image load, 2304 x 16B
        const float4* src = (const float4*)(x + (size_t)b * 9216);
        float4* dst = (float4*)xs;
        for (int i = tid; i < 2304; i += 256) dst[i] = src[i];
    }
    if (tid < 176) ave_s[tid] = 0.0f;
    if (tid < 8)   zpad[tid] = 0.0f;
    __syncthreads();

    const int lane = tid & 31, wid = tid >> 5;
    const int lrow = lane & 15;
    const int kh = (lane >> 4) << 3;

    for (int t = wid; t < 22; t += 8) {           // 11 m-tiles x 2 n-tiles
        const int mt = t >> 1, nt = t & 1;
        const int m = mt * 16 + lrow;
        const bool mv = m < 165;
        const int mc = mv ? m : 0;
        const int oy = mc / 15, ox = mc - oy * 15;
        const float* rowb = xs + oy * 256 + ox * 4;   // (oy*4)*64 + ox*4
        v8f acc = {0.f,0.f,0.f,0.f,0.f,0.f,0.f,0.f};
        #pragma unroll
        for (int ks = 0; ks < 6; ++ks) {
            const int k0 = ks * 32;
            v16h a;
            #pragma unroll
            for (int half = 0; half < 2; ++half) {
                const int kb = k0 + kh + half * 16;   // multiple of 8
                const int c  = kb >> 6;
                const int ky = (kb >> 3) & 7;
                const v4f* p = (const v4f*)(mv ? (rowb + c * 3072 + ky * 64) : zpad);
                v4f va = p[0], vb = p[1];             // 8 consecutive floats
                #pragma unroll
                for (int j = 0; j < 4; ++j) {
                    a[half * 8 + j]     = (_Float16)va[j];
                    a[half * 8 + 4 + j] = (_Float16)vb[j];
                }
            }
            v16h bf = load_bfrag(w1h + (size_t)(nt * 16 + lrow) * 192 + k0 + kh);
            acc = wmma_f16(a, bf, acc);
        }
        const int n = nt * 16 + lrow;
        const float bias = b1[n];
        #pragma unroll
        for (int r = 0; r < 8; ++r) {
            const int mm = mt * 16 + r + kh;          // C row = r + 8*(lane>>4)
            float d = acc[r] + bias;                  // pre-ReLU
            float s = d;                              // channel-sum over 16 lanes
            s += __shfl_xor(s, 1); s += __shfl_xor(s, 2);
            s += __shfl_xor(s, 4); s += __shfl_xor(s, 8);
            if (mm < 165) {
                if (lrow == 0) atomicAdd(&ave_s[mm], s);
                float rv = d > 0.0f ? d : 0.0f;
                x2[(size_t)b * 5280 + (size_t)n * 165 + mm] = (_Float16)rv;
            }
        }
    }
    __syncthreads();
    for (int i = tid; i < 165; i += 256) atomicAdd(&ave1[i], ave_s[i] * (1.0f / 32.0f));
}

// ---------------- conv2: [32,11,15] -(k3 s2)-> [64,5,7], M=35 N=64 K=288 -------
// Build im2col panel As[48][288] in LDS once, then all fragments are ds_load_b128.
__global__ __launch_bounds__(256) void k_conv2(const _Float16* __restrict__ x2,
                                               const float* __restrict__ b2,
                                               const _Float16* __restrict__ w2h,
                                               _Float16* __restrict__ x4,
                                               float* __restrict__ ave2) {
    __shared__ _Float16 xs[32 * 165];    // 10560 B
    __shared__ _Float16 As[48 * 288];    // 27648 B
    __shared__ float ave_s[48];
    const int b = blockIdx.x, tid = threadIdx.x;
    {
        const uint32_t* src = (const uint32_t*)(x2 + (size_t)b * 5280);
        uint32_t* dst = (uint32_t*)xs;
        for (int i = tid; i < 2640; i += 256) dst[i] = src[i];
    }
    if (tid < 48) ave_s[tid] = 0.0f;
    __syncthreads();

    // im2col build (unconditional loads; value select for pad rows)
    for (int e = tid; e < 48 * 288; e += 256) {
        int m = e / 288, k = e - m * 288;
        int c = k / 9, rr = k - c * 9, ky = rr / 3, kx = rr - ky * 3;
        int mc = (m < 35) ? m : 0;
        int oy = mc / 7, ox = mc - oy * 7;
        _Float16 v = xs[c * 165 + (oy * 2 + ky) * 15 + (ox * 2 + kx)];
        As[e] = (m < 35) ? v : (_Float16)0.0f;
    }
    __syncthreads();

    const int lane = tid & 31, wid = tid >> 5;
    const int lrow = lane & 15;
    const int kh = (lane >> 4) << 3;

    for (int t = wid; t < 12; t += 8) {           // 3 m-tiles x 4 n-tiles
        const int mt = t >> 2, nt = t & 3;
        const _Float16* arow = As + (size_t)(mt * 16 + lrow) * 288 + kh;
        const _Float16* brow = w2h + (size_t)(nt * 16 + lrow) * 288 + kh;
        v8f acc = {0.f,0.f,0.f,0.f,0.f,0.f,0.f,0.f};
        #pragma unroll
        for (int ks = 0; ks < 9; ++ks) {
            v16h a  = load_bfrag(arow + ks * 32);
            v16h bf = load_bfrag(brow + ks * 32);
            acc = wmma_f16(a, bf, acc);
        }
        const int n = nt * 16 + lrow;
        const float bias = b2[n];
        #pragma unroll
        for (int r = 0; r < 8; ++r) {
            const int mm = mt * 16 + r + kh;
            float d = acc[r] + bias;
            float s = d;
            s += __shfl_xor(s, 1); s += __shfl_xor(s, 2);
            s += __shfl_xor(s, 4); s += __shfl_xor(s, 8);
            if (mm < 35) {
                if (lrow == 0) atomicAdd(&ave_s[mm], s);
                float rv = d > 0.0f ? d : 0.0f;
                x4[(size_t)b * 2240 + (size_t)n * 35 + mm] = (_Float16)rv;
            }
        }
    }
    __syncthreads();
    for (int i = tid; i < 35; i += 256) atomicAdd(&ave2[i], ave_s[i] * (1.0f / 64.0f));
}

// ---------------- conv3: [64,5,7] -(k3 s1)-> [64,3,5], M=15 N=64 K=576 ---------
__global__ __launch_bounds__(128) void k_conv3(const _Float16* __restrict__ x4,
                                               const float* __restrict__ b3,
                                               const _Float16* __restrict__ w3h,
                                               _Float16* __restrict__ x6,
                                               float* __restrict__ ave3) {
    __shared__ _Float16 xs[64 * 35];     // 4480 B
    __shared__ _Float16 As[16 * 576];    // 18432 B
    __shared__ float ave_s[16];
    const int b = blockIdx.x, tid = threadIdx.x;
    {
        const uint32_t* src = (const uint32_t*)(x4 + (size_t)b * 2240);
        uint32_t* dst = (uint32_t*)xs;
        for (int i = tid; i < 1120; i += 128) dst[i] = src[i];
    }
    if (tid < 16) ave_s[tid] = 0.0f;
    __syncthreads();

    for (int e = tid; e < 16 * 576; e += 128) {
        int m = e / 576, k = e - m * 576;
        int c = k / 9, rr = k - c * 9, ky = rr / 3, kx = rr - ky * 3;
        int mc = (m < 15) ? m : 0;
        int oy = mc / 5, ox = mc - oy * 5;
        _Float16 v = xs[c * 35 + (oy + ky) * 7 + (ox + kx)];
        As[e] = (m < 15) ? v : (_Float16)0.0f;
    }
    __syncthreads();

    const int lane = tid & 31, wid = tid >> 5;    // 4 waves -> 4 n-tiles
    const int lrow = lane & 15;
    const int kh = (lane >> 4) << 3;

    const int nt = wid;
    const _Float16* arow = As + (size_t)lrow * 576 + kh;
    const _Float16* brow = w3h + (size_t)(nt * 16 + lrow) * 576 + kh;
    v8f acc = {0.f,0.f,0.f,0.f,0.f,0.f,0.f,0.f};
    #pragma unroll
    for (int ks = 0; ks < 18; ++ks) {
        v16h a  = load_bfrag(arow + ks * 32);
        v16h bf = load_bfrag(brow + ks * 32);
        acc = wmma_f16(a, bf, acc);
    }
    const int n = nt * 16 + lrow;
    const float bias = b3[n];
    #pragma unroll
    for (int r = 0; r < 8; ++r) {
        const int mm = r + kh;
        float d = acc[r] + bias;
        float s = d;
        s += __shfl_xor(s, 1); s += __shfl_xor(s, 2);
        s += __shfl_xor(s, 4); s += __shfl_xor(s, 8);
        if (mm < 15) {
            if (lrow == 0) atomicAdd(&ave_s[mm], s);
            float rv = d > 0.0f ? d : 0.0f;
            // flatten [C,H,W]: index = n*15 + mm
            x6[(size_t)b * 960 + (size_t)n * 15 + mm] = (_Float16)rv;
        }
    }
    __syncthreads();
    for (int i = tid; i < 15; i += 128) atomicAdd(&ave3[i], ave_s[i] * (1.0f / 64.0f));
}

// ---------------- fc1: [4096,960] x [512,960]^T + b4, ReLU -> h [4096,512] -----
__global__ __launch_bounds__(256) void k_fc1(const _Float16* __restrict__ x6,
                                             const float* __restrict__ b4,
                                             const _Float16* __restrict__ w4h,
                                             _Float16* __restrict__ h) {
    const int mt = blockIdx.x;                    // 0..255
    const int wid = threadIdx.x >> 5, lane = threadIdx.x & 31;
    const int nt = blockIdx.y * 8 + wid;          // 0..31
    const int lrow = lane & 15;
    const int kh = (lane >> 4) << 3;

    const _Float16* arow = x6 + (size_t)(mt * 16 + lrow) * 960 + kh;
    const _Float16* brow = w4h + (size_t)(nt * 16 + lrow) * 960 + kh;
    v8f acc = {0.f,0.f,0.f,0.f,0.f,0.f,0.f,0.f};
    #pragma unroll
    for (int k0 = 0; k0 < 960; k0 += 32) {
        v16h a  = load_bfrag(arow + k0);
        v16h bf = load_bfrag(brow + k0);
        acc = wmma_f16(a, bf, acc);
    }
    const int n = nt * 16 + lrow;
    const float bias = b4[n];
    #pragma unroll
    for (int r = 0; r < 8; ++r) {
        const int m = mt * 16 + r + kh;
        float d = acc[r] + bias;
        d = d > 0.0f ? d : 0.0f;
        h[(size_t)m * 512 + n] = (_Float16)d;
    }
}

// ---------------- fc2: out[b,j] = h[b,:] . w5[j,:] + b5[j], j in {0,1} ---------
__global__ __launch_bounds__(256) void k_fc2(const _Float16* __restrict__ h,
                                             const float* __restrict__ w5,
                                             const float* __restrict__ b5,
                                             float* __restrict__ out) {
    const int wid = threadIdx.x >> 5, lane = threadIdx.x & 31;
    const int b = blockIdx.x * 8 + wid;           // grid 512
    const v8h* hp = (const v8h*)(h + (size_t)b * 512 + lane * 16);
    v8h h0 = hp[0], h1 = hp[1];
    float a0 = 0.0f, a1 = 0.0f;
    #pragma unroll
    for (int i = 0; i < 8; ++i) {
        float v0 = (float)h0[i], v1 = (float)h1[i];
        a0 += v0 * w5[lane * 16 + i]       + v1 * w5[lane * 16 + 8 + i];
        a1 += v0 * w5[512 + lane * 16 + i] + v1 * w5[512 + lane * 16 + 8 + i];
    }
    #pragma unroll
    for (int msk = 1; msk < 32; msk <<= 1) {
        a0 += __shfl_xor(a0, msk);
        a1 += __shfl_xor(a1, msk);
    }
    if (lane == 0) {
        out[(size_t)b * 2 + 0] = a0 + b5[0];
        out[(size_t)b * 2 + 1] = a1 + b5[1];
    }
}

extern "C" void kernel_launch(void* const* d_in, const int* in_sizes, int n_in,
                              void* d_out, int out_size, void* d_ws, size_t ws_size,
                              hipStream_t stream) {
    const float* x  = (const float*)d_in[0];
    const float* w1 = (const float*)d_in[1];
    const float* b1 = (const float*)d_in[2];
    const float* w2 = (const float*)d_in[3];
    const float* b2 = (const float*)d_in[4];
    const float* w3 = (const float*)d_in[5];
    const float* b3 = (const float*)d_in[6];
    const float* w4 = (const float*)d_in[7];
    const float* b4 = (const float*)d_in[8];
    const float* w5 = (const float*)d_in[9];
    const float* b5 = (const float*)d_in[10];
    float* out = (float*)d_out;
    _Float16* wsh = (_Float16*)d_ws;

    k_init<<<1920, 256, 0, stream>>>(w1, w2, w3, w4, wsh, out);
    k_conv1<<<4096, 256, 0, stream>>>(x, b1, wsh + W1_OFF, wsh + X2_OFF, out + 8192);
    k_conv2<<<4096, 256, 0, stream>>>(wsh + X2_OFF, b2, wsh + W2_OFF, wsh + X4_OFF, out + 8357);
    k_conv3<<<4096, 128, 0, stream>>>(wsh + X4_OFF, b3, wsh + W3_OFF, wsh + X6_OFF, out + 8392);
    k_fc1<<<dim3(256, 4), 256, 0, stream>>>(wsh + X6_OFF, b4, wsh + W4_OFF, wsh + H_OFF);
    k_fc2<<<512, 256, 0, stream>>>(wsh + H_OFF, w5, b5, out);
}
